// D2STGNN_59021440582353
// MI455X (gfx1250) — matrix operand order repaired and checked by reading, not compiled, not run on hardware
//
#include <hip/hip_runtime.h>

// ---------------------------------------------------------------------------
// D2STGNN forward for MI455X (gfx1250, wave32, WMMA).
// All GEMMs use v_wmma_f32_16x16x32_bf16 (bf16 operands, f32 accumulate),
// register-blocked so each scattered A fragment feeds 2-4 WMMAs.
// ---------------------------------------------------------------------------

typedef __attribute__((ext_vector_type(16))) __bf16 v16bf;
typedef __attribute__((ext_vector_type(8)))  float  v8f;

namespace {
constexpr int kB   = 4;
constexpr int kL   = 12;
constexpr int kN   = 1024;
constexpr int kC   = 3;
constexpr int kD   = 32;
constexpr int kND  = 32;
constexpr int kTE  = 10;
constexpr int kST  = 96;
constexpr int kGH  = 64;
constexpr int kTID = 288;
constexpr int kDIW = 7;
constexpr int kCOLS = kL * kD;                    // 384
constexpr int kROWS = kB * kN * kL;               // 49152
constexpr long long kCHUNK = (long long)kB * kN * kCOLS;  // 1572864 elems per chunk

// workspace layout (bytes)
constexpr size_t OFF_DINV_F = 0;
constexpr size_t OFF_DINV_B = OFF_DINV_F + (size_t)kB * kN * 4;
constexpr size_t OFF_STAT   = OFF_DINV_B + (size_t)kB * kN * 4;
constexpr size_t OFF_TAB_T  = OFF_STAT   + (size_t)kN * kN * 4;
constexpr size_t OFF_TAB_D  = OFF_TAB_T  + (size_t)kTID * kGH * 4;
constexpr size_t OFF_GNODE  = OFF_TAB_D  + (size_t)kDIW * kGH * 4;
constexpr size_t OFF_HST    = OFF_GNODE  + (size_t)kN * kGH * 4;
constexpr size_t OFF_CHUNKS = OFF_HST    + (size_t)kROWS * kD * 4;
constexpr size_t WS_NEEDED  = OFF_CHUNKS + (size_t)7 * kCHUNK * 2;
}

__device__ __forceinline__ __bf16 tobf(float x) { return (__bf16)x; }

// A-fragment K index for 16-bit A 16x32: lane half h in {0,1},
// element i in [0,16): K = i + 16h (contiguous, matching B fills).
__device__ __forceinline__ int a_k_of(int i, int h) {
  return i + 16 * h;
}

__device__ __forceinline__ v8f wmma_bf16(v16bf a, v16bf b, v8f c) {
  return __builtin_amdgcn_wmma_f32_16x16x32_bf16(false, a, false, b, (short)0, c,
                                                 false, false);
}

// ---------------------------------------------------------------------------
// 1) Degree normalizers for random walks: 1/(1 + rowsum), 1/(1 + colsum)
// ---------------------------------------------------------------------------
__global__ void deg_kernel(const float* __restrict__ adj,
                           float* __restrict__ dinv_f,
                           float* __restrict__ dinv_b) {
  int idx = blockIdx.x * blockDim.x + threadIdx.x;
  if (idx >= kB * kN) return;
  int b = idx / kN, n = idx % kN;
  const float* Ab = adj + (size_t)b * kN * kN;
  float sr = 1.0f;  // +1 from identity
  float sc = 1.0f;
  for (int m = 0; m < kN; ++m) {
    sr += Ab[(size_t)n * kN + m];
    sc += Ab[(size_t)m * kN + n];
  }
  dinv_f[idx] = 1.0f / sr;
  dinv_b[idx] = 1.0f / sc;
}

// ---------------------------------------------------------------------------
// 2) static logits: relu(node_emb_d @ node_emb_u^T), 1024x1024, K=32.
//    16x64 tile per wave: one A fragment feeds 4 WMMAs.
// ---------------------------------------------------------------------------
__global__ void static_logits_wmma(const float* __restrict__ emb_d,
                                   const float* __restrict__ emb_u,
                                   float* __restrict__ stat) {
  int tm = blockIdx.x;             // 64 row tiles
  int tn = blockIdx.y;             // 16 tiles of 64 columns
  int lane = threadIdx.x, h = lane >> 4, ln = lane & 15;
  int row = tm * 16 + ln;
  v16bf a;
#pragma unroll
  for (int i = 0; i < 16; ++i) a[i] = tobf(emb_d[row * kND + a_k_of(i, h)]);

  v8f acc[4];
#pragma unroll
  for (int t = 0; t < 4; ++t) {
    int col = tn * 64 + t * 16 + ln;   // B col = emb_u row (B = emb_u^T)
    v16bf bb;
#pragma unroll
    for (int i = 0; i < 16; ++i) bb[i] = tobf(emb_u[col * kND + (i + 16 * h)]);
    v8f z = {};
    acc[t] = wmma_bf16(a, bb, z);
  }
#pragma unroll
  for (int t = 0; t < 4; ++t) {
#pragma unroll
    for (int r = 0; r < 8; ++r) {
      int m = tm * 16 + r + 8 * h;
      stat[(size_t)m * kN + tn * 64 + t * 16 + ln] = fmaxf(acc[t][r], 0.0f);
    }
  }
}

// ---------------------------------------------------------------------------
// 3) row softmax over stat (axis=1), one wave per row
// ---------------------------------------------------------------------------
__global__ void softmax_rows(float* __restrict__ stat) {
  int n = blockIdx.x;
  int lane = threadIdx.x;
  float* row = stat + (size_t)n * kN;
  float mx = -1e30f;
  for (int m = lane; m < kN; m += 32) mx = fmaxf(mx, row[m]);
  for (int o = 16; o > 0; o >>= 1) mx = fmaxf(mx, __shfl_xor(mx, o, 32));
  float s = 0.0f;
  for (int m = lane; m < kN; m += 32) s += __expf(row[m] - mx);
  for (int o = 16; o > 0; o >>= 1) s += __shfl_xor(s, o, 32);
  float inv = 1.0f / s;
  for (int m = lane; m < kN; m += 32) row[m] = __expf(row[m] - mx) * inv;
}

// ---------------------------------------------------------------------------
// 4) gate factorization tables: tab_t=TiD@Wg1[0:10], tab_d=DiW@Wg1[10:20],
//    g_node = eu@Wg1[20:52] + ed@Wg1[52:84] + b_g1
// ---------------------------------------------------------------------------
__global__ void gate_tables(const float* __restrict__ TiD,
                            const float* __restrict__ DiW,
                            const float* __restrict__ eu,
                            const float* __restrict__ ed,
                            const float* __restrict__ Wg1,
                            const float* __restrict__ bg1,
                            float* __restrict__ tab_t,
                            float* __restrict__ tab_d,
                            float* __restrict__ g_node) {
  int idx = blockIdx.x * blockDim.x + threadIdx.x;
  if (idx >= (kTID + kDIW + kN) * kGH) return;
  int h = idx % kGH, r = idx / kGH;
  if (r < kTID) {
    float s = 0.0f;
    for (int e = 0; e < kTE; ++e) s += TiD[r * kTE + e] * Wg1[e * kGH + h];
    tab_t[r * kGH + h] = s;
  } else if (r < kTID + kDIW) {
    int rr = r - kTID;
    float s = 0.0f;
    for (int e = 0; e < kTE; ++e) s += DiW[rr * kTE + e] * Wg1[(kTE + e) * kGH + h];
    tab_d[rr * kGH + h] = s;
  } else {
    int nn = r - kTID - kDIW;
    float s = bg1[h];
    for (int k = 0; k < kND; ++k) s += eu[nn * kND + k] * Wg1[(2 * kTE + k) * kGH + h];
    for (int k = 0; k < kND; ++k) s += ed[nn * kND + k] * Wg1[(2 * kTE + kND + k) * kGH + h];
    g_node[nn * kGH + h] = s;
  }
}

// ---------------------------------------------------------------------------
// 5) H_st = hidden_states_st @ W_st : (49152 x 96) @ (96 x 32), K=3x32.
//    Full 32-col output per wave: one A fragment feeds 2 WMMAs per K step.
// ---------------------------------------------------------------------------
__global__ void hst_wmma(const float* __restrict__ hst,
                         const float* __restrict__ Wst,
                         float* __restrict__ Hout) {
  int tm = blockIdx.x;
  int lane = threadIdx.x, h = lane >> 4, ln = lane & 15;
  int row = tm * 16 + ln;
  const float* Arow = hst + (size_t)row * kST;
  v8f acc[2] = {{}, {}};
  for (int kb = 0; kb < kST; kb += 32) {
    v16bf a;
#pragma unroll
    for (int i = 0; i < 16; ++i) a[i] = tobf(Arow[kb + a_k_of(i, h)]);
#pragma unroll
    for (int t = 0; t < 2; ++t) {
      v16bf bb;
#pragma unroll
      for (int i = 0; i < 16; ++i)
        bb[i] = tobf(Wst[(kb + i + 16 * h) * kD + t * 16 + ln]);
      acc[t] = wmma_bf16(a, bb, acc[t]);
    }
  }
#pragma unroll
  for (int t = 0; t < 2; ++t) {
#pragma unroll
    for (int r = 0; r < 8; ++r) {
      int m = tm * 16 + r + 8 * h;
      Hout[(size_t)m * kD + t * 16 + ln] = acc[t][r];
    }
  }
}

// ---------------------------------------------------------------------------
// 6) fuse: gate MLP (factorized) + embed + relu + gate -> chunk0 (bf16)
//    one thread per (b,n,l) position
// ---------------------------------------------------------------------------
__global__ void fuse_xspa(const float* __restrict__ hist,
                          const float* __restrict__ Hst,
                          const float* __restrict__ tab_t,
                          const float* __restrict__ tab_d,
                          const float* __restrict__ g_node,
                          const float* __restrict__ Wg2,
                          const float* __restrict__ bg2,
                          const float* __restrict__ Wemb,
                          const float* __restrict__ bemb,
                          const float* __restrict__ bst,
                          __bf16* __restrict__ chunk0) {
  int idx = blockIdx.x * blockDim.x + threadIdx.x;
  if (idx >= kB * kN * kL) return;
  int l = idx % kL;
  int bn = idx / kL;
  int n = bn % kN;
  int b = bn / kN;
  const float* hd = hist + (((size_t)(b * kL + l)) * kN + n) * kC;
  float x0 = hd[0];
  int t_idx = (int)(hd[1] * (float)kTID);
  int d_idx = (int)((hd[2] * (1.0f / 7.0f)) * (float)kDIW);
  const float* tt = tab_t + t_idx * kGH;
  const float* td = tab_d + d_idx * kGH;
  const float* gn = g_node + n * kGH;
  float logit = bg2[0];
#pragma unroll 8
  for (int h2 = 0; h2 < kGH; ++h2)
    logit += fmaxf(tt[h2] + td[h2] + gn[h2], 0.0f) * Wg2[h2];
  float gate = 1.0f / (1.0f + __expf(-logit));
  const float* hrow = Hst + (size_t)idx * kD;
  __bf16* orow = chunk0 + (size_t)idx * kD;  // (B,N,L,D) == (B,N,COLS)
#pragma unroll 8
  for (int d = 0; d < kD; ++d) {
    float v = x0 * Wemb[d] + bemb[d] + hrow[d] + bst[d];
    orow[d] = tobf(fmaxf(v, 0.0f) * gate);
  }
}

// ---------------------------------------------------------------------------
// 7) propagation GEMM: out[b][n][col] = norm * sum_m A(n,m) * Z[b][m][col]
//    mode 0: A = static softmax (shared over b)
//    mode 1: A = adj[b] + I, scale rows by dinv_f   (rw_fwd)
//    mode 2: A = adj[b]^T + I, scale rows by dinv_b (rw_bwd)
//    16x64 tile per wave, K=1024: each A fragment feeds 4 WMMAs (128 total).
// ---------------------------------------------------------------------------
__global__ void prop_wmma(const float* __restrict__ adj,
                          const float* __restrict__ stat,
                          const float* __restrict__ dinv,
                          const __bf16* __restrict__ Zin,
                          __bf16* __restrict__ Zout,
                          int mode) {
  int tm = blockIdx.x;   // 64 n tiles
  int tc = blockIdx.y;   // 6 tiles of 64 columns
  int b = blockIdx.z;
  int lane = threadIdx.x, h = lane >> 4, ln = lane & 15;
  int nrow = tm * 16 + ln;

  const float* Asrc;
  size_t rstride, cstride;
  if (mode == 0)      { Asrc = stat;                          rstride = kN; cstride = 1;  }
  else if (mode == 1) { Asrc = adj + (size_t)b * kN * kN;     rstride = kN; cstride = 1;  }
  else                { Asrc = adj + (size_t)b * kN * kN;     rstride = 1;  cstride = kN; }
  const float* Abase = Asrc + (size_t)nrow * rstride;
  const __bf16* Zb = Zin + (size_t)b * kN * kCOLS + tc * 64 + ln;

  v8f acc[4] = {{}, {}, {}, {}};
  for (int kb = 0; kb < kN; kb += 32) {
    if (kb + 32 < kN) {  // prefetch next K slab of A (global_prefetch_b8)
      __builtin_prefetch(Abase + (size_t)(kb + 32) * cstride, 0, 3);
    }
    v16bf a;
#pragma unroll
    for (int i = 0; i < 16; ++i) {
      int k = kb + a_k_of(i, h);
      float v = Abase[(size_t)k * cstride];
      if (mode != 0 && k == nrow) v += 1.0f;  // + identity
      a[i] = tobf(v);
    }
#pragma unroll
    for (int t = 0; t < 4; ++t) {
      v16bf zfrag;
#pragma unroll
      for (int i = 0; i < 16; ++i) {
        int k = kb + i + 16 * h;
        zfrag[i] = Zb[(size_t)k * kCOLS + t * 16];
      }
      acc[t] = wmma_bf16(a, zfrag, acc[t]);
    }
  }
  __bf16* Ob = Zout + (size_t)b * kN * kCOLS;
#pragma unroll
  for (int r = 0; r < 8; ++r) {
    int m = tm * 16 + r + 8 * h;
    float norm = (mode != 0) ? dinv[b * kN + m] : 1.0f;
#pragma unroll
    for (int t = 0; t < 4; ++t) {
      Ob[(size_t)m * kCOLS + tc * 64 + t * 16 + ln] = tobf(acc[t][r] * norm);
    }
  }
}

// ---------------------------------------------------------------------------
// 8) final: out = Hcat(7x32 bf16 chunks) @ W_dif + b_dif, K=224.
//    Full 32-col output per wave: one A fragment feeds 2 WMMAs per chunk.
//    Scatter to (B,L,N,D) output order, f32.
// ---------------------------------------------------------------------------
__global__ void final_wmma(const __bf16* __restrict__ chunks,
                           const float* __restrict__ Wdif,
                           const float* __restrict__ bdif,
                           float* __restrict__ out) {
  int tm = blockIdx.x;  // row tile over (b,n,l)
  int lane = threadIdx.x, h = lane >> 4, ln = lane & 15;
  int pos = tm * 16 + ln;              // (b*N+n)*L + l
  size_t base = (size_t)pos * kD;      // == bn*COLS + l*32
  v8f acc[2] = {{}, {}};
  for (int c = 0; c < 7; ++c) {
    const __bf16* ch = chunks + (size_t)c * kCHUNK + base;
    v16bf a;
#pragma unroll
    for (int i = 0; i < 16; ++i) a[i] = ch[a_k_of(i, h)];
#pragma unroll
    for (int t = 0; t < 2; ++t) {
      v16bf bb;
#pragma unroll
      for (int i = 0; i < 16; ++i)
        bb[i] = tobf(Wdif[(c * 32 + i + 16 * h) * kD + t * 16 + ln]);
      acc[t] = wmma_bf16(a, bb, acc[t]);
    }
  }
#pragma unroll
  for (int r = 0; r < 8; ++r) {
    int m = tm * 16 + r + 8 * h;       // (b*N+n)*L + l
    int l2 = m % kL;
    int bn2 = m / kL;
    int n2 = bn2 % kN;
    int b2 = bn2 / kN;
    float* orow = out + (((size_t)(b2 * kL + l2)) * kN + n2) * kD;
#pragma unroll
    for (int t = 0; t < 2; ++t) {
      orow[t * 16 + ln] = acc[t][r] + bdif[t * 16 + ln];
    }
  }
}

// ---------------------------------------------------------------------------
// host launcher
// ---------------------------------------------------------------------------
extern "C" void kernel_launch(void* const* d_in, const int* in_sizes, int n_in,
                              void* d_out, int out_size, void* d_ws, size_t ws_size,
                              hipStream_t stream) {
  (void)in_sizes; (void)n_in; (void)out_size;
  if (ws_size < WS_NEEDED) return;

  const float* hist = (const float*)d_in[0];   // (B,L,N,C)
  const float* hst  = (const float*)d_in[1];   // (B,N,L,ST)
  const float* adj  = (const float*)d_in[2];   // (B,N,N)
  const float* Wemb = (const float*)d_in[3];
  const float* bemb = (const float*)d_in[4];
  const float* Wst  = (const float*)d_in[5];
  const float* bst  = (const float*)d_in[6];
  const float* TiD  = (const float*)d_in[7];
  const float* DiW  = (const float*)d_in[8];
  const float* eu   = (const float*)d_in[9];
  const float* ed   = (const float*)d_in[10];
  const float* Wg1  = (const float*)d_in[11];
  const float* bg1  = (const float*)d_in[12];
  const float* Wg2  = (const float*)d_in[13];
  const float* bg2  = (const float*)d_in[14];
  const float* Wdif = (const float*)d_in[15];
  const float* bdif = (const float*)d_in[16];

  char* ws = (char*)d_ws;
  float*  dinv_f = (float*)(ws + OFF_DINV_F);
  float*  dinv_b = (float*)(ws + OFF_DINV_B);
  float*  statp  = (float*)(ws + OFF_STAT);
  float*  tab_t  = (float*)(ws + OFF_TAB_T);
  float*  tab_d  = (float*)(ws + OFF_TAB_D);
  float*  g_node = (float*)(ws + OFF_GNODE);
  float*  Hst    = (float*)(ws + OFF_HST);
  __bf16* chunks = (__bf16*)(ws + OFF_CHUNKS);

  // 1) random-walk degree normalizers
  deg_kernel<<<(kB * kN + 255) / 256, 256, 0, stream>>>(adj, dinv_f, dinv_b);

  // 2) static adjacency: relu(emb_d @ emb_u^T) then row softmax
  static_logits_wmma<<<dim3(kN / 16, kN / 64), 32, 0, stream>>>(ed, eu, statp);
  softmax_rows<<<kN, 32, 0, stream>>>(statp);

  // 3) gate factorization tables
  {
    int total = (kTID + kDIW + kN) * kGH;
    gate_tables<<<(total + 255) / 256, 256, 0, stream>>>(TiD, DiW, eu, ed, Wg1, bg1,
                                                         tab_t, tab_d, g_node);
  }

  // 4) H_st GEMM
  hst_wmma<<<kROWS / 16, 32, 0, stream>>>(hst, Wst, Hst);

  // 5) fused embed + relu + gate -> chunk 0 (X_spa, bf16)
  fuse_xspa<<<(kB * kN * kL + 255) / 256, 256, 0, stream>>>(
      hist, Hst, tab_t, tab_d, g_node, Wg2, bg2, Wemb, bemb, bst, chunks);

  // 6) diffusion: supports in reference order [static, rw_fwd, rw_bwd], 2 hops each
  dim3 pgrid(kN / 16, kCOLS / 64, kB);
  for (int s = 0; s < 3; ++s) {
    const float* dinv = (s == 2) ? dinv_b : dinv_f;
    __bf16* z1 = chunks + (size_t)(1 + 2 * s) * kCHUNK;
    __bf16* z2 = chunks + (size_t)(2 + 2 * s) * kCHUNK;
    prop_wmma<<<pgrid, 32, 0, stream>>>(adj, statp, dinv, chunks, z1, s);
    prop_wmma<<<pgrid, 32, 0, stream>>>(adj, statp, dinv, z1, z2, s);
  }

  // 7) output projection: Hcat(224) @ W_dif + b_dif -> (B,L,N,D) f32
  final_wmma<<<kROWS / 16, 32, 0, stream>>>(chunks, Wdif, bdif, (float*)d_out);
}
